// QuaternionPointLinear_76836964926173
// MI455X (gfx1250) — matrix operand order repaired and checked by reading, not compiled, not run on hardware
//
#include <hip/hip_runtime.h>

// ---------------------------------------------------------------------------
// Quaternion pointwise linear == one dense GEMM:
//   Out[M,512] = X[M,512] x Wc[512,512],  M = 8*16384 = 131072
// Roofline: 68.7 GFLOP vs 537 MB HBM traffic (X + Out) -> memory bound
// (~23us @ 23.3 TB/s) when matrix cores run bf16 (v_wmma_f32_16x16x32_bf16,
// fp32 accumulate). Combined weight is 512 KB bf16 -> L2-resident, B loaded
// straight from global. A converted fp32->bf16 with the native packed
// converter and staged through double-buffered LDS (1 barrier per K-step).
// ---------------------------------------------------------------------------

typedef __attribute__((ext_vector_type(16))) __bf16        v16bf;
typedef __attribute__((ext_vector_type(4)))  __bf16        v4bf;
typedef __attribute__((ext_vector_type(4)))  float         v4f;
typedef __attribute__((ext_vector_type(8)))  float         v8f;
typedef __attribute__((ext_vector_type(8)))  unsigned int  v8u;

// ---------------------------------------------------------------------------
// Build combined 512x512 bf16 weight, stored "N-major":
//   Wn[oc_global * 512 + ic_global]
// Hamilton product table:
//   out_r = +Wr x_r - Wi x_i - Wj x_j - Wk x_k
//   out_i = +Wi x_r + Wr x_i + Wk x_j - Wj x_k
//   out_j = +Wj x_r - Wk x_i + Wr x_j + Wi x_k
//   out_k = +Wk x_r + Wj x_i - Wi x_j + Wr x_k
// ---------------------------------------------------------------------------
__global__ __launch_bounds__(256) void build_wc_kernel(
    const float* __restrict__ wr, const float* __restrict__ wi,
    const float* __restrict__ wj, const float* __restrict__ wk,
    __bf16* __restrict__ Wn)
{
  const int idx  = blockIdx.x * 256 + threadIdx.x;   // 0 .. 512*512-1
  const int oc_g = idx >> 9;
  const int ic_g = idx & 511;
  const int ocomp = oc_g >> 7, oc = oc_g & 127;
  const int icomp = ic_g >> 7, ic = ic_g & 127;

  const int   type_tab[4][4] = {{0,1,2,3},{1,0,3,2},{2,3,0,1},{3,2,1,0}};
  const float sign_tab[4][4] = {{1.f,-1.f,-1.f,-1.f},
                                {1.f, 1.f, 1.f,-1.f},
                                {1.f,-1.f, 1.f, 1.f},
                                {1.f, 1.f,-1.f, 1.f}};
  const float* ws[4] = {wr, wi, wj, wk};

  const float v = sign_tab[ocomp][icomp] * ws[type_tab[ocomp][icomp]][oc * 128 + ic];
  Wn[idx] = (__bf16)v;   // RTNE via hardware converter
}

// ---------------------------------------------------------------------------
// Main GEMM. Block tile 128(M) x 128(N), K-step 32. 256 threads = 8 waves,
// arranged 4(M) x 2(N); each wave owns a 32x64 tile = 2x4 WMMA accumulators.
// A (fp32) -> bf16 via v_cvt_pk_bf16_f32, staged in double-buffered LDS
// (row stride 80 B: 16B-aligned ds_load_b128, conflict-free fragment reads).
// ---------------------------------------------------------------------------
#define BM  128
#define BN  128
#define ALD 40   // __bf16 per LDS row (32 data + 8 pad = 80 bytes)

__global__ __launch_bounds__(256) void qgemm_kernel(
    const float* __restrict__ X, const __bf16* __restrict__ Wn,
    const float* __restrict__ bias, float* __restrict__ Out, int M)
{
  __shared__ __bf16 Alds[2][BM * ALD];   // 2 x 10 KB

  const int tid  = threadIdx.x;
  const int lane = tid & 31;
  const int wave = tid >> 5;
  const int wm   = wave >> 1;       // 0..3  (M direction)
  const int wn   = wave & 1;        // 0..1  (N direction)
  const int l16  = lane & 15;
  const int half = lane >> 4;

  const int  n0 = blockIdx.x * BN;
  const long m0 = (long)blockIdx.y * BM;

  // A-staging: each thread loads 4x float4 (16B chunk in each of 4 rows)
  const int arow = tid >> 3;        // 0..31
  const int acol = (tid & 7) << 2;  // 0,4,...,28

  const float* __restrict__ Xblk = X + m0 * 512;

  v4f areg[4];
#pragma unroll
  for (int p = 0; p < 4; ++p)
    areg[p] = *(const v4f*)(Xblk + (size_t)(arow + 32 * p) * 512 + acol);

  v8f acc[2][4] = {};   // zero-initialized accumulators

  for (int kt = 0; kt < 16; ++kt) {
    const int k0  = kt * 32;
    const int buf = kt & 1;

    // ---- stage A tile (fp32 -> bf16 packed convert, ds_store_b64)
#pragma unroll
    for (int p = 0; p < 4; ++p) {
      const v4bf b = __builtin_convertvector(areg[p], v4bf);
      *(v4bf*)(&Alds[buf][(arow + 32 * p) * ALD + acol]) = b;
    }

    // Single barrier per K-step: the previous iteration's barrier already
    // guarantees this buffer's old readers are done (double buffering).
    __syncthreads();

    // ---- prefetch next A tile while this one is consumed
    if (kt < 15) {
#pragma unroll
      for (int p = 0; p < 4; ++p)
        areg[p] = *(const v4f*)(Xblk + (size_t)(arow + 32 * p) * 512 +
                                (k0 + 32) + acol);
    }

    // ---- A fragments from LDS (ISA 16-bit A layout: K = 16*(v>=4)+8*half+2*(v&3))
    v16bf afrag[2];
#pragma unroll
    for (int mf = 0; mf < 2; ++mf) {
      const __bf16* rp = &Alds[buf][(wm * 32 + mf * 16 + l16) * ALD];
      const uint4 r0 = *(const uint4*)(rp + half * 8);        // K = 8h .. 8h+7
      const uint4 r1 = *(const uint4*)(rp + 16 + half * 8);   // K = 16+8h ..
      const v8u u = {r0.x, r0.y, r0.z, r0.w, r1.x, r1.y, r1.z, r1.w};
      afrag[mf] = __builtin_bit_cast(v16bf, u);
    }

    // ---- B fragments straight from global/L2 (same K order as A: K = 16*(v>=4)+8*half+2*(v&3))
    v16bf bfrag[4];
#pragma unroll
    for (int nf = 0; nf < 4; ++nf) {
      const int n = n0 + wn * 64 + nf * 16 + l16;
      const uint4* bp = (const uint4*)(Wn + (size_t)n * 512 + k0 + half * 8);
      const uint4 b0 = bp[0];
      const uint4 b1 = bp[2];
      const v8u u = {b0.x, b0.y, b0.z, b0.w, b1.x, b1.y, b1.z, b1.w};
      bfrag[nf] = __builtin_bit_cast(v16bf, u);
    }

    // ---- 8 WMMAs per K-step
#pragma unroll
    for (int mf = 0; mf < 2; ++mf)
#pragma unroll
      for (int nf = 0; nf < 4; ++nf)
        acc[mf][nf] = __builtin_amdgcn_wmma_f32_16x16x32_bf16(
            false, afrag[mf], false, bfrag[nf],
            (short)0, acc[mf][nf], false, false);
  }

  // ---- Epilogue: C layout row M = r + 8*half, col N = lane%16; add bias.
#pragma unroll
  for (int mf = 0; mf < 2; ++mf) {
#pragma unroll
    for (int nf = 0; nf < 4; ++nf) {
      const int   n  = n0 + wn * 64 + nf * 16 + l16;
      const float bv = bias[n];
#pragma unroll
      for (int r = 0; r < 8; ++r) {
        const long m = m0 + wm * 32 + mf * 16 + half * 8 + r;
        Out[m * 512 + n] = acc[mf][nf][r] + bv;
      }
    }
  }
}

// ---------------------------------------------------------------------------
extern "C" void kernel_launch(void* const* d_in, const int* in_sizes, int n_in,
                              void* d_out, int out_size, void* d_ws, size_t ws_size,
                              hipStream_t stream) {
  const float* x    = (const float*)d_in[0];
  const float* wr   = (const float*)d_in[1];
  const float* wi   = (const float*)d_in[2];
  const float* wj   = (const float*)d_in[3];
  const float* wk   = (const float*)d_in[4];
  const float* bias = (const float*)d_in[5];
  float* out = (float*)d_out;

  const int M = in_sizes[0] / 512;          // 131072 for the reference
  __bf16* Wn = (__bf16*)d_ws;               // 512*512*2 = 512 KB scratch

  // 1) assemble combined quaternion weight matrix (bf16, N-major)
  build_wc_kernel<<<(512 * 512) / 256, 256, 0, stream>>>(wr, wi, wj, wk, Wn);

  // 2) the GEMM: grid = (N/128, M/128)
  dim3 grid(512 / BN, (unsigned)(M / BM));
  qgemm_kernel<<<grid, 256, 0, stream>>>(x, Wn, bias, out, M);
}
